// TransformerBlock_81630148428467
// MI455X (gfx1250) — matrix operand order repaired and checked by reading, not compiled, not run on hardware
//
#include <hip/hip_runtime.h>

// ---------------------------------------------------------------------------
// Types for CDNA5 WMMA (wave32): bf16 16x16x32, f32 accumulate
// ---------------------------------------------------------------------------
typedef __attribute__((ext_vector_type(16))) __bf16 v16bf;
typedef __attribute__((ext_vector_type(8)))  __bf16 v8bf;
typedef __attribute__((ext_vector_type(8)))  float  v8f;

union uv16 { v16bf v; v8bf h[2]; };

#define WMMA_BF16(A, B, C) \
    __builtin_amdgcn_wmma_f32_16x16x32_bf16(false, (A), false, (B), (short)0, (C), false, false)

// Problem constants (match the reference)
#define BATCH 2
#define SLEN  2048
#define DMODEL 1024
#define NHEAD 16
#define HDIM  64
#define FFDIM 4096
#define MROWS (BATCH * SLEN)   // 4096

// ---------------------------------------------------------------------------
// Weight transpose + fp32 -> bf16:  Wt[n*Kd + k] = W[k*Nd + n]
// ---------------------------------------------------------------------------
__global__ __launch_bounds__(256)
void k_transpose_bf16(const float* __restrict__ W, __bf16* __restrict__ Wt,
                      int Kd, int Nd)
{
    int idx = blockIdx.x * 256 + threadIdx.x;
    if (idx >= Kd * Nd) return;
    int n = idx / Kd;
    int k = idx - n * Kd;
    Wt[idx] = (__bf16)W[(size_t)k * Nd + n];
}

// ---------------------------------------------------------------------------
// LayerNorm (torch semantics: (x - mean) / (std(ddof=1) + eps) * g + s),
// one 256-thread block per row of 1024, bf16 output for GEMM A operand.
// ---------------------------------------------------------------------------
__global__ __launch_bounds__(256)
void k_layernorm_bf16(const float* __restrict__ x, const float* __restrict__ g,
                      const float* __restrict__ sh, __bf16* __restrict__ y)
{
    __shared__ float rs[8], rq[8];
    int row = blockIdx.x;
    int t = threadIdx.x;
    const float* xr = x + (size_t)row * DMODEL;
    float4 v = *(const float4*)(xr + t * 4);
    float a[4] = {v.x, v.y, v.z, v.w};
    float s = a[0] + a[1] + a[2] + a[3];
    float q = a[0]*a[0] + a[1]*a[1] + a[2]*a[2] + a[3]*a[3];
    #pragma unroll
    for (int off = 16; off >= 1; off >>= 1) {
        s += __shfl_xor(s, off, 32);
        q += __shfl_xor(q, off, 32);
    }
    if ((t & 31) == 0) { rs[t >> 5] = s; rq[t >> 5] = q; }
    __syncthreads();
    float ts = 0.f, tq = 0.f;
    #pragma unroll
    for (int i = 0; i < 8; ++i) { ts += rs[i]; tq += rq[i]; }
    float mean = ts * (1.0f / DMODEL);
    float var  = fmaxf((tq - (float)DMODEL * mean * mean) * (1.0f / (DMODEL - 1)), 0.0f);
    float inv  = 1.0f / (sqrtf(var) + 1e-5f);
    __bf16* yr = y + (size_t)row * DMODEL;
    #pragma unroll
    for (int e = 0; e < 4; ++e) {
        int d = t * 4 + e;
        yr[d] = (__bf16)((a[e] - mean) * inv * g[d] + sh[d]);
    }
}

// ---------------------------------------------------------------------------
// Generic bf16 WMMA GEMM: C[M,N] = A[M,K] @ Bt[N,K]^T + bias, epilogue modes.
// Block = 256 threads (8 waves), tile 128x64; wave w -> rows [w*16, w*16+16),
// 4 column tiles of 16. K stepped by 32.
//   epi 0: QKV scatter  -> outH as [B, H, S, HD] bf16
//   epi 1: O-proj       -> outF = acc + bias + resid            (fp32 [M,N])
//   epi 2: FFN1         -> outH = bf16(gelu_exact(acc + bias))
//   epi 3: FFN2         -> outF = acc + bias + resid            (fp32 [M,N])
// ---------------------------------------------------------------------------
__global__ __launch_bounds__(256)
void k_gemm_bf16(const __bf16* __restrict__ A, const __bf16* __restrict__ Bt,
                 const float* __restrict__ bias, const float* __restrict__ resid,
                 float* __restrict__ outF, __bf16* __restrict__ outH,
                 int M, int N, int K, int epi)
{
    const int wave = threadIdx.x >> 5;
    const int lane = threadIdx.x & 31;
    const int half = lane >> 4;
    const int l15  = lane & 15;
    const int m0 = blockIdx.x * 128 + wave * 16;
    const int n0 = blockIdx.y * 64;

    v8f acc[4] = {};
    const __bf16* arow = A + (size_t)(m0 + l15) * K + half * 16;

    for (int kk = 0; kk < K; kk += 32) {
        uv16 af;
        af.h[0] = *(const v8bf*)(arow + kk);        // k = kk + half*16 + [0,8)
        af.h[1] = *(const v8bf*)(arow + kk + 8);    // k = kk + half*16 + 8 + [0,8)
        #pragma unroll
        for (int nt = 0; nt < 4; ++nt) {
            const __bf16* bp = Bt + (size_t)(n0 + nt * 16 + l15) * K + kk + half * 16;
            v16bf bv = *(const v16bf*)bp;           // 16 contiguous K values
            acc[nt] = WMMA_BF16(af.v, bv, acc[nt]);
        }
    }

    #pragma unroll
    for (int nt = 0; nt < 4; ++nt) {
        int gn = n0 + nt * 16 + l15;
        float bsv = bias[gn];
        #pragma unroll
        for (int r = 0; r < 8; ++r) {
            int gm = m0 + half * 8 + r;
            float c = acc[nt][r] + bsv;
            if (epi == 0) {
                // scatter to [b, h, s, hd]
                int b  = gm >> 11;          // / SLEN
                int si = gm & (SLEN - 1);
                int hh = gn >> 6;           // / HDIM
                int hd = gn & (HDIM - 1);
                outH[(((size_t)(b * NHEAD + hh) * SLEN + si) * HDIM) + hd] = (__bf16)c;
            } else if (epi == 1) {
                outF[(size_t)gm * N + gn] = c + resid[(size_t)gm * N + gn];
            } else if (epi == 2) {
                float gl = 0.5f * c * (1.0f + erff(c * 0.70710678118f));
                outH[(size_t)gm * N + gn] = (__bf16)gl;
            } else {
                outF[(size_t)gm * N + gn] = c + resid[(size_t)gm * N + gn];
            }
        }
    }
}

// ---------------------------------------------------------------------------
// Flash attention (causal), per block: one (b,h), one 64-row Q tile.
// 128 threads = 4 waves; wave w owns 16 query rows. K/V tiles of 64 keys
// staged in LDS (V transposed so PV B-fragments are contiguous).
// ---------------------------------------------------------------------------
__global__ __launch_bounds__(128)
void k_attn_flash(const __bf16* __restrict__ Qh, const __bf16* __restrict__ Kh,
                  const __bf16* __restrict__ Vh, __bf16* __restrict__ ctx)
{
    __shared__ __bf16 Kl[64][72];      // [key][hd], padded
    __shared__ __bf16 Vt[64][72];      // [hd][key], padded
    __shared__ __bf16 Pl[4][16][64];   // per-wave P tile

    const int wave = threadIdx.x >> 5;
    const int lane = threadIdx.x & 31;
    const int half = lane >> 4;
    const int l15  = lane & 15;
    const int qt = blockIdx.x;         // q tile (64 rows)
    const int bh = blockIdx.y;         // b*NHEAD + h
    const int qbase = qt * 64;

    const __bf16* Qb = Qh + (size_t)bh * SLEN * HDIM;
    const __bf16* Kb = Kh + (size_t)bh * SLEN * HDIM;
    const __bf16* Vb = Vh + (size_t)bh * SLEN * HDIM;

    // Persistent Q fragments (K-dim 64 -> two WMMA steps)
    const __bf16* qp = Qb + (size_t)(qbase + wave * 16 + l15) * HDIM + half * 16;
    uv16 qf[2];
    qf[0].h[0] = *(const v8bf*)(qp);
    qf[0].h[1] = *(const v8bf*)(qp + 8);
    qf[1].h[0] = *(const v8bf*)(qp + 32);
    qf[1].h[1] = *(const v8bf*)(qp + 40);

    v8f o[4] = {};
    float mrow[8], lrow[8];
    #pragma unroll
    for (int r = 0; r < 8; ++r) { mrow[r] = -1e30f; lrow[r] = 0.0f; }

    for (int kt = 0; kt <= qt; ++kt) {
        const int kbase = kt * 64;
        // cooperative LDS fill: 64x64 bf16 tiles, 8-element vectors
        #pragma unroll
        for (int i = 0; i < 4; ++i) {
            int linear = threadIdx.x + i * 128;       // 0..511
            int row = linear >> 3;
            int col = (linear & 7) * 8;
            v8bf kv = *(const v8bf*)(Kb + (size_t)(kbase + row) * HDIM + col);
            *(v8bf*)(&Kl[row][col]) = kv;
            v8bf vv = *(const v8bf*)(Vb + (size_t)(kbase + row) * HDIM + col);
            #pragma unroll
            for (int e = 0; e < 8; ++e) Vt[col + e][row] = vv[e];
        }
        __syncthreads();

        // S = Q @ K^T  (B operand: Kl[key][hd] rows are contiguous)
        v8f sc[4] = {};
        #pragma unroll
        for (int nt = 0; nt < 4; ++nt) {
            const __bf16* bp0 = &Kl[nt * 16 + l15][half * 16];
            uv16 b0; b0.h[0] = *(const v8bf*)bp0; b0.h[1] = *(const v8bf*)(bp0 + 8);
            sc[nt] = WMMA_BF16(qf[0].v, b0.v, sc[nt]);
            const __bf16* bp1 = &Kl[nt * 16 + l15][32 + half * 16];
            uv16 b1; b1.h[0] = *(const v8bf*)bp1; b1.h[1] = *(const v8bf*)(bp1 + 8);
            sc[nt] = WMMA_BF16(qf[1].v, b1.v, sc[nt]);
        }

        // online softmax, causal mask on diagonal tile only
        const bool diag = (kt == qt);
        #pragma unroll
        for (int r = 0; r < 8; ++r) {
            int q_abs = qbase + wave * 16 + half * 8 + r;
            float vals[4];
            float rmax = -1e30f;
            #pragma unroll
            for (int nt = 0; nt < 4; ++nt) {
                float vv = sc[nt][r] * 0.125f;       // 1/sqrt(64)
                int k_abs = kbase + nt * 16 + l15;
                if (diag && k_abs > q_abs) vv = -100000.0f;
                vals[nt] = vv;
                rmax = fmaxf(rmax, vv);
            }
            #pragma unroll
            for (int off = 1; off < 16; off <<= 1)
                rmax = fmaxf(rmax, __shfl_xor(rmax, off, 32));
            float mnew = fmaxf(mrow[r], rmax);
            float corr = __expf(mrow[r] - mnew);
            float rsum = 0.0f;
            #pragma unroll
            for (int nt = 0; nt < 4; ++nt) {
                float p = __expf(vals[nt] - mnew);
                rsum += p;
                Pl[wave][half * 8 + r][nt * 16 + l15] = (__bf16)p;
            }
            #pragma unroll
            for (int off = 1; off < 16; off <<= 1)
                rsum += __shfl_xor(rsum, off, 32);
            lrow[r] = lrow[r] * corr + rsum;
            mrow[r] = mnew;
            #pragma unroll
            for (int nt = 0; nt < 4; ++nt) o[nt][r] *= corr;
        }

        // O += P @ V   (A: Pl per-wave; B: Vt[hd][key] rows contiguous)
        const __bf16* prow = &Pl[wave][l15][0];
        #pragma unroll
        for (int ks = 0; ks < 2; ++ks) {
            uv16 af;
            af.h[0] = *(const v8bf*)(prow + ks * 32 + half * 16);
            af.h[1] = *(const v8bf*)(prow + ks * 32 + half * 16 + 8);
            #pragma unroll
            for (int nt = 0; nt < 4; ++nt) {
                const __bf16* vp = &Vt[nt * 16 + l15][ks * 32 + half * 16];
                uv16 bf_; bf_.h[0] = *(const v8bf*)vp; bf_.h[1] = *(const v8bf*)(vp + 8);
                o[nt] = WMMA_BF16(af.v, bf_.v, o[nt]);
            }
        }
        __syncthreads();
    }

    // normalize, write ctx back to [b, s, d] layout (bf16 for O-proj GEMM)
    const int b  = bh >> 4;
    const int hh = bh & 15;
    #pragma unroll
    for (int nt = 0; nt < 4; ++nt) {
        #pragma unroll
        for (int r = 0; r < 8; ++r) {
            int si = qbase + wave * 16 + half * 8 + r;
            int d  = hh * HDIM + nt * 16 + l15;
            float val = o[nt][r] / lrow[r];
            ctx[((size_t)(b * SLEN + si)) * DMODEL + d] = (__bf16)val;
        }
    }
}

// ---------------------------------------------------------------------------
// Host orchestration
// ---------------------------------------------------------------------------
extern "C" void kernel_launch(void* const* d_in, const int* in_sizes, int n_in,
                              void* d_out, int out_size, void* d_ws, size_t ws_size,
                              hipStream_t stream)
{
    (void)in_sizes; (void)n_in; (void)out_size; (void)ws_size;

    const float* x  = (const float*)d_in[0];
    const float* Wq = (const float*)d_in[1];  const float* bq = (const float*)d_in[2];
    const float* Wk = (const float*)d_in[3];  const float* bk = (const float*)d_in[4];
    const float* Wv = (const float*)d_in[5];  const float* bv = (const float*)d_in[6];
    const float* Wo = (const float*)d_in[7];  const float* bo = (const float*)d_in[8];
    const float* W1 = (const float*)d_in[9];  const float* b1 = (const float*)d_in[10];
    const float* W2 = (const float*)d_in[11]; const float* b2 = (const float*)d_in[12];
    const float* g1 = (const float*)d_in[13]; const float* s1 = (const float*)d_in[14];
    const float* g2 = (const float*)d_in[15]; const float* s2 = (const float*)d_in[16];

    char* wsp = (char*)d_ws;
    auto take = [&](size_t bytes) -> char* {
        char* p = wsp;
        wsp += (bytes + 255) & ~(size_t)255;
        return p;
    };

    const size_t DD  = (size_t)DMODEL * DMODEL;          // 1M
    const size_t DF  = (size_t)DMODEL * FFDIM;           // 4M
    const size_t MD  = (size_t)MROWS * DMODEL;           // 4M
    const size_t MF  = (size_t)MROWS * FFDIM;            // 16M

    __bf16* Wqt = (__bf16*)take(DD * 2);
    __bf16* Wkt = (__bf16*)take(DD * 2);
    __bf16* Wvt = (__bf16*)take(DD * 2);
    __bf16* Wot = (__bf16*)take(DD * 2);
    __bf16* W1t = (__bf16*)take(DF * 2);   // [FFDIM][DMODEL]
    __bf16* W2t = (__bf16*)take(DF * 2);   // [DMODEL][FFDIM]
    __bf16* h1  = (__bf16*)take(MD * 2);   // LN1 out
    __bf16* qb  = (__bf16*)take(MD * 2);   // [B,H,S,HD]
    __bf16* kb  = (__bf16*)take(MD * 2);
    __bf16* vb  = (__bf16*)take(MD * 2);
    __bf16* ctx = (__bf16*)take(MD * 2);   // [B,S,D]
    float*  x2  = (float*)take(MD * 4);    // residual 1 out
    __bf16* h2  = (__bf16*)take(MD * 2);   // LN2 out
    __bf16* f1  = (__bf16*)take(MF * 2);   // gelu(FFN1) out

    // 1) weights -> bf16, transposed [N][K]
    k_transpose_bf16<<<(int)((DD + 255) / 256), 256, 0, stream>>>(Wq, Wqt, DMODEL, DMODEL);
    k_transpose_bf16<<<(int)((DD + 255) / 256), 256, 0, stream>>>(Wk, Wkt, DMODEL, DMODEL);
    k_transpose_bf16<<<(int)((DD + 255) / 256), 256, 0, stream>>>(Wv, Wvt, DMODEL, DMODEL);
    k_transpose_bf16<<<(int)((DD + 255) / 256), 256, 0, stream>>>(Wo, Wot, DMODEL, DMODEL);
    k_transpose_bf16<<<(int)((DF + 255) / 256), 256, 0, stream>>>(W1, W1t, DMODEL, FFDIM);
    k_transpose_bf16<<<(int)((DF + 255) / 256), 256, 0, stream>>>(W2, W2t, FFDIM, DMODEL);

    // 2) LN1
    k_layernorm_bf16<<<MROWS, 256, 0, stream>>>(x, g1, s1, h1);

    // 3) QKV projections (epi 0: scatter to [B,H,S,HD])
    dim3 gDD(MROWS / 128, DMODEL / 64);
    k_gemm_bf16<<<gDD, 256, 0, stream>>>(h1, Wqt, bq, nullptr, nullptr, qb,
                                         MROWS, DMODEL, DMODEL, 0);
    k_gemm_bf16<<<gDD, 256, 0, stream>>>(h1, Wkt, bk, nullptr, nullptr, kb,
                                         MROWS, DMODEL, DMODEL, 0);
    k_gemm_bf16<<<gDD, 256, 0, stream>>>(h1, Wvt, bv, nullptr, nullptr, vb,
                                         MROWS, DMODEL, DMODEL, 0);

    // 4) causal flash attention
    k_attn_flash<<<dim3(SLEN / 64, BATCH * NHEAD), 128, 0, stream>>>(qb, kb, vb, ctx);

    // 5) output projection + residual (x)  -> x2 fp32
    k_gemm_bf16<<<gDD, 256, 0, stream>>>(ctx, Wot, bo, x, x2, nullptr,
                                         MROWS, DMODEL, DMODEL, 1);

    // 6) LN2
    k_layernorm_bf16<<<MROWS, 256, 0, stream>>>(x2, g2, s2, h2);

    // 7) FFN1 + exact GELU -> f1 bf16
    dim3 gDF(MROWS / 128, FFDIM / 64);
    k_gemm_bf16<<<gDF, 256, 0, stream>>>(h2, W1t, b1, nullptr, nullptr, f1,
                                         MROWS, FFDIM, DMODEL, 2);

    // 8) FFN2 + residual (x2) -> d_out fp32
    k_gemm_bf16<<<gDD, 256, 0, stream>>>(f1, W2t, b2, x2, (float*)d_out, nullptr,
                                         MROWS, DMODEL, FFDIM, 3);
}
